// OCAB_15539191677367
// MI455X (gfx1250) — compile-verified, matrix operand order checked by reading, not run on hardware
//
#include <hip/hip_runtime.h>
#include <hip/hip_bf16.h>
#include <math.h>

typedef __attribute__((ext_vector_type(16))) __bf16 v16bf;
typedef __attribute__((ext_vector_type(8)))  float  v8f;
typedef __attribute__((ext_vector_type(4)))  unsigned v4u;
typedef __attribute__((ext_vector_type(8)))  int   v8i;
typedef __attribute__((ext_vector_type(4)))  int   v4i;

#define ROWS   65536   // 4 * 128 * 128
#define CDIM   192
#define NHEAD  6
#define HDIM   32
#define HID    384

#if !__has_builtin(__builtin_amdgcn_tensor_load_to_lds)
#define NO_TDM 1
#endif

__device__ __forceinline__ __bf16 f2bf(float f) {
    union { float f; unsigned u; } x; x.f = f;
    unsigned r = (x.u + 0x7FFFu + ((x.u >> 16) & 1u)) >> 16;
    union { unsigned short s; __bf16 b; } y; y.s = (unsigned short)r;
    return y.b;
}

__device__ __forceinline__ v8f wmma_bf16(v16bf a, v16bf b, v8f c) {
    return __builtin_amdgcn_wmma_f32_16x16x32_bf16(
        false, a, false, b, (short)0, c, false, false);
}

// LDS byte offset of a __shared__ object: flat addr low 32 bits == LDS offset
__device__ __forceinline__ unsigned lds_off(const void* p) {
    return (unsigned)(size_t)p;
}

#ifndef NO_TDM
// 2D TDM load: tile_d1 rows x tile_d0 elems of 2-byte data, row stride = stride0
// elems, packed contiguously into LDS at lds_byte. Built per ISA 8.3/8.4.
__device__ __forceinline__ void tdm_load_2d(
    unsigned lds_byte, const void* gptr,
    unsigned tensor_d0, unsigned tensor_d1,
    unsigned tile_d0, unsigned tile_d1, unsigned stride0) {
    unsigned long long ga = (unsigned long long)(size_t)gptr;
    v4u g0;
    g0[0] = 1u;                                             // count=1, user mode
    g0[1] = lds_byte;                                       // lds_addr
    g0[2] = (unsigned)(ga & 0xFFFFFFFFu);                   // global_addr lo
    g0[3] = (unsigned)((ga >> 32) & 0x01FFFFFFu) | 0x80000000u; // hi | type=2
    v8i g1;
    g1[0] = (int)(1u << 16);                                // data_size=1 (2B)
    g1[1] = (int)((tensor_d0 & 0xFFFFu) << 16);             // dim0[15:0]
    g1[2] = (int)((tensor_d0 >> 16) | ((tensor_d1 & 0xFFFFu) << 16));
    g1[3] = (int)((tensor_d1 >> 16) | (tile_d0 << 16));
    g1[4] = (int)tile_d1;                                   // tile_dim1, dim2=0
    g1[5] = (int)stride0;                                   // dim0_stride lo
    g1[6] = 0;
    g1[7] = 0;
    v4i z4; z4[0] = 0; z4[1] = 0; z4[2] = 0; z4[3] = 0;
#if __clang_major__ >= 23
    v8i z8; z8[0]=0; z8[1]=0; z8[2]=0; z8[3]=0; z8[4]=0; z8[5]=0; z8[6]=0; z8[7]=0;
    __builtin_amdgcn_tensor_load_to_lds(g0, g1, z4, z4, z8, 0);
#else
    __builtin_amdgcn_tensor_load_to_lds(g0, g1, z4, z4, 0);
#endif
}
#endif

// ---------------------------------------------------------------------------
// Weight convert + transpose: W[K][N] fp32 -> Wt[N][K] bf16
// ---------------------------------------------------------------------------
__global__ void wt_kernel(const float* __restrict__ w, __bf16* __restrict__ wt,
                          int K, int N) {
    int i = blockIdx.x * 256 + threadIdx.x;
    if (i >= N * K) return;
    int n = i / K, k = i - n * K;
    wt[i] = f2bf(w[(size_t)k * N + n]);
}

// ---------------------------------------------------------------------------
// LayerNorm over C=192, one wave per row, 8 rows per block
// ---------------------------------------------------------------------------
__global__ __launch_bounds__(256) void ln_kernel(
    const float* __restrict__ x, const float* __restrict__ g,
    const float* __restrict__ b, __bf16* __restrict__ out) {
    int row  = blockIdx.x * 8 + (threadIdx.x >> 5);
    int lane = threadIdx.x & 31;
    const float* xr = x + (size_t)row * CDIM;
    float v[6]; float s = 0.f;
    #pragma unroll
    for (int i = 0; i < 6; ++i) { v[i] = xr[lane + 32 * i]; s += v[i]; }
    #pragma unroll
    for (int m = 16; m > 0; m >>= 1) s += __shfl_xor(s, m, 32);
    float mu = s * (1.0f / CDIM);
    float vs = 0.f;
    #pragma unroll
    for (int i = 0; i < 6; ++i) { float d = v[i] - mu; vs += d * d; }
    #pragma unroll
    for (int m = 16; m > 0; m >>= 1) vs += __shfl_xor(vs, m, 32);
    float rstd = rsqrtf(vs * (1.0f / CDIM) + 1e-5f);
    __bf16* orow = out + (size_t)row * CDIM;
    #pragma unroll
    for (int i = 0; i < 6; ++i) {
        int c = lane + 32 * i;
        orow[c] = f2bf((v[i] - mu) * rstd * g[c] + b[c]);
    }
}

// ---------------------------------------------------------------------------
// WMMA GEMM: C[M,N] = A[M,K](bf16) @ Bt[N,K]^T(bf16) + bias (+res)(+gelu)
// 256 thr = 8 waves (4x2), block tile 128x64, wave tile 32x32, K-step 32.
// Tiles staged via TDM (tensor_load_to_lds) with LDS double buffering.
// ---------------------------------------------------------------------------
template <bool STORE_BF16, bool DO_GELU, bool DO_RES>
__global__ __launch_bounds__(256) void gemm_bf16_kernel(
    const __bf16* __restrict__ A, const __bf16* __restrict__ Bt,
    const float* __restrict__ bias, const float* __restrict__ res,
    void* __restrict__ Cout, int M, int N, int K) {
    __shared__ __attribute__((aligned(32))) __bf16 Asl[2][128 * 32];
    __shared__ __attribute__((aligned(32))) __bf16 Bsl[2][64 * 32];
    int tid = threadIdx.x;
    int wave = tid >> 5, lane = tid & 31;
    int wm = wave >> 1, wn = wave & 1;
    int lane_lo = lane & 15, khalf = (lane >> 4) * 16, half8 = (lane >> 4) * 8;
    int bm = blockIdx.x * 128, bn = blockIdx.y * 64;

    v8f acc[2][2];
    #pragma unroll
    for (int mt = 0; mt < 2; ++mt)
        #pragma unroll
        for (int nt = 0; nt < 2; ++nt)
            #pragma unroll
            for (int j = 0; j < 8; ++j) acc[mt][nt][j] = 0.0f;

    int nk = K / 32;
#ifndef NO_TDM
    bool issuer = (wave == 0);
    if (issuer) {
        tdm_load_2d(lds_off(&Asl[0][0]), A + (size_t)bm * K, K, M, 32, 128, K);
        tdm_load_2d(lds_off(&Bsl[0][0]), Bt + (size_t)bn * K, K, N, 32, 64, K);
    }
#endif
    for (int i = 0; i < nk; ++i) {
        int buf = i & 1;
#ifndef NO_TDM
        if (issuer) {
            if (i + 1 < nk) {   // prefetch next K-tile into alternate buffer
                tdm_load_2d(lds_off(&Asl[buf ^ 1][0]),
                            A + (size_t)bm * K + (i + 1) * 32, K, M, 32, 128, K);
                tdm_load_2d(lds_off(&Bsl[buf ^ 1][0]),
                            Bt + (size_t)bn * K + (i + 1) * 32, K, N, 32, 64, K);
                __builtin_amdgcn_s_wait_tensorcnt(2);  // current buf complete
            } else {
                __builtin_amdgcn_s_wait_tensorcnt(0);
            }
        }
#else
        {   // manual staging fallback
            int r = tid >> 1, ch = (tid & 1) * 16;
            *(v16bf*)&Asl[buf][r * 32 + ch] =
                *(const v16bf*)(A + (size_t)(bm + r) * K + i * 32 + ch);
            int r2 = tid >> 2, c8 = (tid & 3) * 8;
            *(uint4*)&Bsl[buf][r2 * 32 + c8] =
                *(const uint4*)(Bt + (size_t)(bn + r2) * K + i * 32 + c8);
        }
#endif
        __syncthreads();
        #pragma unroll
        for (int mt = 0; mt < 2; ++mt) {
            v16bf af = *(const v16bf*)&Asl[buf][(wm * 32 + mt * 16 + lane_lo) * 32 + khalf];
            #pragma unroll
            for (int nt = 0; nt < 2; ++nt) {
                v16bf bf = *(const v16bf*)&Bsl[buf][(wn * 32 + nt * 16 + lane_lo) * 32 + khalf];
                acc[mt][nt] = wmma_bf16(af, bf, acc[mt][nt]);
            }
        }
        __syncthreads();
    }

    #pragma unroll
    for (int mt = 0; mt < 2; ++mt)
        #pragma unroll
        for (int nt = 0; nt < 2; ++nt) {
            int col = bn + wn * 32 + nt * 16 + lane_lo;
            float bv = bias[col];
            #pragma unroll
            for (int j = 0; j < 8; ++j) {
                int row = bm + wm * 32 + mt * 16 + half8 + j;
                float v = acc[mt][nt][j] + bv;
                if (DO_GELU) v = 0.5f * v * (1.0f + erff(v * 0.70710678118654752f));
                if (DO_RES)  v += res[(size_t)row * N + col];
                if (STORE_BF16) ((__bf16*)Cout)[(size_t)row * N + col] = f2bf(v);
                else            ((float*)Cout)[(size_t)row * N + col] = v;
            }
        }
}

// ---------------------------------------------------------------------------
// Fused overlapping-window attention (flash-style), one block per (window,head)
// m-subtiles processed sequentially to keep live S-tile at 32 VGPRs.
// ---------------------------------------------------------------------------
__global__ __launch_bounds__(256) void attn_kernel(
    const __bf16* __restrict__ qkv,   // [ROWS][576]
    const float*  __restrict__ rpb,   // [1521][6]
    __bf16* __restrict__ o) {         // [ROWS][192]
    __shared__ __attribute__((aligned(32))) __bf16 Ks[64 * 32];
    __shared__ __attribute__((aligned(32))) __bf16 Vts[32 * 64];
    __shared__ __attribute__((aligned(32))) __bf16 Ps[8][32 * 64];

    int head = blockIdx.x % NHEAD;
    int win  = blockIdx.x / NHEAD;
    int wb = win >> 6, wy = (win & 63) >> 3, wx = win & 7;
    int tid = threadIdx.x, wave = tid >> 5, lane = tid & 31;
    int lane_lo = lane & 15, khalf = (lane >> 4) * 16, half8 = (lane >> 4) * 8;
    int qbase = wave * 32;

    v16bf qf[2];
    #pragma unroll
    for (int mt = 0; mt < 2; ++mt) {
        int qrow = qbase + mt * 16 + lane_lo;
        int qi = qrow >> 4, qj = qrow & 15;
        size_t grow = (size_t)wb * 16384 + (size_t)(wy * 16 + qi) * 128 + (wx * 16 + qj);
        qf[mt] = *(const v16bf*)(qkv + grow * 576 + head * HDIM + khalf);
    }

    v8f oacc[2][2];
    float mrun[2][8], lrun[2][8];
    #pragma unroll
    for (int mt = 0; mt < 2; ++mt) {
        #pragma unroll
        for (int nt = 0; nt < 2; ++nt)
            #pragma unroll
            for (int j = 0; j < 8; ++j) oacc[mt][nt][j] = 0.0f;
        #pragma unroll
        for (int j = 0; j < 8; ++j) { mrun[mt][j] = -1e30f; lrun[mt][j] = 0.0f; }
    }

    const float scale = 0.17677669529663687f;  // 1/sqrt(32)

    for (int c = 0; c < 9; ++c) {
        {   // stage K (natural) and V (transposed) chunk in LDS
            int kl = tid >> 2;
            int h8 = (tid & 3) * 8;
            int key = c * 64 + kl;
            int oy = key / 24, ox = key - oy * 24;
            int gy = wy * 16 - 4 + oy, gx = wx * 16 - 4 + ox;
            uint4 kq, vq;
            if (gy >= 0 && gy < 128 && gx >= 0 && gx < 128) {
                size_t grow = (size_t)wb * 16384 + (size_t)gy * 128 + gx;
                const __bf16* kp = qkv + grow * 576 + CDIM + head * HDIM + h8;
                kq = *(const uint4*)kp;
                vq = *(const uint4*)(kp + CDIM);
            } else {
                kq = make_uint4(0, 0, 0, 0);
                vq = make_uint4(0, 0, 0, 0);
            }
            *(uint4*)&Ks[kl * 32 + h8] = kq;
            const __bf16* vvals = (const __bf16*)&vq;
            #pragma unroll
            for (int j = 0; j < 8; ++j) Vts[(h8 + j) * 64 + kl] = vvals[j];
        }
        __syncthreads();

        // key-side rel-bias offsets: idx = (qi*39+qj+920) - (oy*39+ox)
        int koff[4];
        #pragma unroll
        for (int nt = 0; nt < 4; ++nt) {
            int key = c * 64 + nt * 16 + lane_lo;
            int oy = key / 24;
            koff[nt] = oy * 39 + (key - oy * 24);
        }

        // ---- S = Q @ K^T + softmax, one m-subtile at a time (32-VGPR S) ----
        #pragma unroll
        for (int mt = 0; mt < 2; ++mt) {
            v8f sacc[4];
            #pragma unroll
            for (int nt = 0; nt < 4; ++nt)
                #pragma unroll
                for (int j = 0; j < 8; ++j) sacc[nt][j] = 0.0f;
            #pragma unroll
            for (int nt = 0; nt < 4; ++nt) {
                v16bf kf = *(const v16bf*)&Ks[(nt * 16 + lane_lo) * 32 + khalf];
                sacc[nt] = wmma_bf16(qf[mt], kf, sacc[nt]);
            }
            #pragma unroll
            for (int j = 0; j < 8; ++j) {
                int qrow = qbase + mt * 16 + half8 + j;
                int qoff = ((qrow >> 4) * 39) + (qrow & 15) + 920;
                float rmax = -1e30f;
                #pragma unroll
                for (int nt = 0; nt < 4; ++nt) {
                    float sv = sacc[nt][j] * scale +
                               rpb[(qoff - koff[nt]) * NHEAD + head];
                    sacc[nt][j] = sv;
                    rmax = fmaxf(rmax, sv);
                }
                #pragma unroll
                for (int m = 8; m > 0; m >>= 1)
                    rmax = fmaxf(rmax, __shfl_xor(rmax, m, 16));
                float mnew  = fmaxf(mrun[mt][j], rmax);
                float alpha = __expf(mrun[mt][j] - mnew);
                float rsum = 0.0f;
                #pragma unroll
                for (int nt = 0; nt < 4; ++nt) {
                    float p = __expf(sacc[nt][j] - mnew);
                    sacc[nt][j] = p;
                    rsum += p;
                }
                #pragma unroll
                for (int m = 8; m > 0; m >>= 1)
                    rsum += __shfl_xor(rsum, m, 16);
                lrun[mt][j] = lrun[mt][j] * alpha + rsum;
                mrun[mt][j] = mnew;
                oacc[mt][0][j] *= alpha;
                oacc[mt][1][j] *= alpha;
            }
            #pragma unroll
            for (int nt = 0; nt < 4; ++nt)
                #pragma unroll
                for (int j = 0; j < 8; ++j)
                    Ps[wave][(mt * 16 + half8 + j) * 64 + nt * 16 + lane_lo] =
                        f2bf(sacc[nt][j]);
        }

        // ---- O += P @ V ----
        #pragma unroll
        for (int ks = 0; ks < 2; ++ks) {
            v16bf pf[2];
            #pragma unroll
            for (int mt = 0; mt < 2; ++mt)
                pf[mt] = *(const v16bf*)&Ps[wave][(mt * 16 + lane_lo) * 64 + ks * 32 + khalf];
            #pragma unroll
            for (int nt = 0; nt < 2; ++nt) {
                v16bf vf = *(const v16bf*)&Vts[(nt * 16 + lane_lo) * 64 + ks * 32 + khalf];
                #pragma unroll
                for (int mt = 0; mt < 2; ++mt)
                    oacc[mt][nt] = wmma_bf16(pf[mt], vf, oacc[mt][nt]);
            }
        }
        __syncthreads();
    }

    #pragma unroll
    for (int mt = 0; mt < 2; ++mt)
        #pragma unroll
        for (int j = 0; j < 8; ++j) {
            float inv = 1.0f / lrun[mt][j];
            int qrow = qbase + mt * 16 + half8 + j;
            int qi = qrow >> 4, qj = qrow & 15;
            size_t grow = (size_t)wb * 16384 + (size_t)(wy * 16 + qi) * 128 + (wx * 16 + qj);
            #pragma unroll
            for (int nt = 0; nt < 2; ++nt) {
                int hd = nt * 16 + lane_lo;
                o[grow * CDIM + head * HDIM + hd] = f2bf(oacc[mt][nt][j] * inv);
            }
        }
}

// ---------------------------------------------------------------------------
// Launcher
// ---------------------------------------------------------------------------
extern "C" void kernel_launch(void* const* d_in, const int* in_sizes, int n_in,
                              void* d_out, int out_size, void* d_ws, size_t ws_size,
                              hipStream_t stream) {
    const float* x      = (const float*)d_in[0];
    const float* n1g    = (const float*)d_in[1];
    const float* n1b    = (const float*)d_in[2];
    const float* qkv_w  = (const float*)d_in[3];
    const float* qkv_b  = (const float*)d_in[4];
    const float* rpb    = (const float*)d_in[5];
    const float* proj_w = (const float*)d_in[6];
    const float* proj_b = (const float*)d_in[7];
    const float* n2g    = (const float*)d_in[8];
    const float* n2b    = (const float*)d_in[9];
    const float* fc1_w  = (const float*)d_in[10];
    const float* fc1_b  = (const float*)d_in[11];
    const float* fc2_w  = (const float*)d_in[12];
    const float* fc2_b  = (const float*)d_in[13];

    char* ws = (char*)d_ws;
    size_t off = 0;
    __bf16* xn    = (__bf16*)(ws + off); off += (size_t)ROWS * CDIM * 2;
    __bf16* qkvb  = (__bf16*)(ws + off); off += (size_t)ROWS * 576  * 2;
    __bf16* ob    = (__bf16*)(ws + off); off += (size_t)ROWS * CDIM * 2;
    float*  x2    = (float*) (ws + off); off += (size_t)ROWS * CDIM * 4;
    __bf16* qkvwt = (__bf16*)(ws + off); off += (size_t)576 * CDIM * 2;
    __bf16* projwt= (__bf16*)(ws + off); off += (size_t)CDIM * CDIM * 2;
    __bf16* fc1wt = (__bf16*)(ws + off); off += (size_t)HID * CDIM * 2;
    __bf16* fc2wt = (__bf16*)(ws + off); off += (size_t)CDIM * HID * 2;
    __bf16* yb = xn;    // reuse (xn dead after QKV GEMM)
    __bf16* hb = qkvb;  // reuse (qkv dead after attention)

    wt_kernel<<<(576 * CDIM + 255) / 256, 256, 0, stream>>>(qkv_w,  qkvwt, CDIM, 576);
    wt_kernel<<<(CDIM * CDIM + 255) / 256, 256, 0, stream>>>(proj_w, projwt, CDIM, CDIM);
    wt_kernel<<<(HID * CDIM + 255) / 256, 256, 0, stream>>>(fc1_w,  fc1wt, CDIM, HID);
    wt_kernel<<<(CDIM * HID + 255) / 256, 256, 0, stream>>>(fc2_w,  fc2wt, HID, CDIM);

    ln_kernel<<<ROWS / 8, 256, 0, stream>>>(x, n1g, n1b, xn);

    gemm_bf16_kernel<true, false, false><<<dim3(ROWS / 128, 576 / 64), 256, 0, stream>>>(
        xn, qkvwt, qkv_b, nullptr, (void*)qkvb, ROWS, 576, CDIM);

    attn_kernel<<<256 * NHEAD, 256, 0, stream>>>(qkvb, rpb, ob);

    gemm_bf16_kernel<false, false, true><<<dim3(ROWS / 128, CDIM / 64), 256, 0, stream>>>(
        ob, projwt, proj_b, x, (void*)x2, ROWS, CDIM, CDIM);

    ln_kernel<<<ROWS / 8, 256, 0, stream>>>(x2, n2g, n2b, yb);

    gemm_bf16_kernel<true, true, false><<<dim3(ROWS / 128, HID / 64), 256, 0, stream>>>(
        yb, fc1wt, fc1_b, nullptr, (void*)hb, ROWS, HID, CDIM);

    gemm_bf16_kernel<false, false, true><<<dim3(ROWS / 128, CDIM / 64), 256, 0, stream>>>(
        hb, fc2wt, fc2_b, x2, d_out, ROWS, CDIM, HID);
}
